// GCNII_84559316124075
// MI455X (gfx1250) — matrix operand order repaired and checked by reading, not compiled, not run on hardware
//
#include <hip/hip_runtime.h>
#include <hip/hip_bf16.h>
#include <cmath>

#define NNODES   50000
#define NEDGES   600000
#define FIN      50
#define HDIM     128
#define FOUT     121
#define NLAYERS  6
#define KPAD_IN  64
#define ALPHA_C  0.1f

typedef __bf16 bf16;
typedef __attribute__((ext_vector_type(16))) __bf16 v16bf;
typedef __attribute__((ext_vector_type(8)))  float  v8f;

union Frag { v16bf v; uint4 q[2]; };

// Build a 16-element bf16 WMMA fragment from two 16-byte chunks.
__device__ __forceinline__ v16bf load_frag(const bf16* p0, const bf16* p1) {
  Frag f;
  f.q[0] = *(const uint4*)p0;
  f.q[1] = *(const uint4*)p1;
  return f.v;
}

__device__ __forceinline__ v8f wmma_bf16f32(v16bf a, v16bf b, v8f c) {
  // D = A(16x32 bf16) x B(32x16 bf16) + C(16x16 f32)
  return __builtin_amdgcn_wmma_f32_16x16x32_bf16(
      /*neg_a=*/false, a, /*neg_b=*/false, b,
      /*c_mod=*/(short)0, c, /*reuse_a=*/false, /*reuse_b=*/false);
}

// -------- weight convert + transpose (f32 [Ksrc,Nsrc] -> bf16 [Npad,Kpad], zero padded)
__global__ void k_cvt_transpose(const float* __restrict__ src, unsigned short* __restrict__ dstu,
                                int Ksrc, int Nsrc, int Kpad, int Npad) {
  bf16* dst = (bf16*)dstu;
  int t = blockIdx.x * blockDim.x + threadIdx.x;
  if (t >= Kpad * Npad) return;
  int k = t % Kpad, n = t / Kpad;
  float v = (k < Ksrc && n < Nsrc) ? src[k * Nsrc + n] : 0.0f;
  dst[n * Kpad + k] = (bf16)v;
}

// -------- zero a float buffer (float4 vectorized)
__global__ void k_zero4(float4* __restrict__ p, int n4) {
  int t = blockIdx.x * blockDim.x + threadIdx.x;
  if (t < n4) p[t] = make_float4(0.f, 0.f, 0.f, 0.f);
}

// -------- SpMM scatter: agg[row] += ew * x[col]; 32 threads (one wave) per edge, 4 feats/thread
__global__ void k_spmm(const long long* __restrict__ rowi, const long long* __restrict__ coli,
                       const float* __restrict__ ew, const float* __restrict__ x,
                       float* __restrict__ agg) {
  int t = blockIdx.x * blockDim.x + threadIdx.x;
  int e = t >> 5;
  if (e >= NEDGES) return;
  int c = (t & 31) << 2;
  int r = (int)rowi[e];
  int s = (int)coli[e];
  float w = ew[e];
  const float4 xv = *(const float4*)(x + s * HDIM + c);
  float* a = agg + r * HDIM + c;
  atomicAdd(a + 0, w * xv.x);
  atomicAdd(a + 1, w * xv.y);
  atomicAdd(a + 2, w * xv.z);
  atomicAdd(a + 3, w * xv.w);
}

// -------- input projection: x0 = x = relu(x_in @ w_in + b_in)   [N,50]@[50,128]
__global__ void __launch_bounds__(128)
k_linear_in(const float* __restrict__ xin, const unsigned short* __restrict__ wTu,
            const float* __restrict__ bias, float* __restrict__ x0, float* __restrict__ x) {
  const bf16* wT = (const bf16*)wTu;
  __shared__ __align__(16) bf16 sA[16 * KPAD_IN];
  int mbase = blockIdx.x * 16;
  int tid = threadIdx.x;
  for (int i = tid; i < 16 * KPAD_IN; i += 128) {
    int m = i / KPAD_IN, k = i % KPAD_IN;
    float v = (k < FIN) ? xin[(mbase + m) * FIN + k] : 0.0f;
    sA[i] = (bf16)v;
  }
  __syncthreads();
  int wave = tid >> 5, lane = tid & 31;
  int hi8  = (lane >= 16) ? 8  : 0;
  int hi16 = (lane >= 16) ? 16 : 0;
  int mrow = lane & 15;
  int n0 = (wave * 2 + 0) * 16 + mrow;
  int n1 = (wave * 2 + 1) * 16 + mrow;
  v8f acc0 = {}; v8f acc1 = {};
#pragma unroll
  for (int ks = 0; ks < KPAD_IN / 32; ++ks) {
    int kb = ks * 32;
    v16bf a  = load_frag(&sA[mrow * KPAD_IN + kb + hi8], &sA[mrow * KPAD_IN + kb + 16 + hi8]);
    v16bf b0 = load_frag(&wT[n0 * KPAD_IN + kb + hi8], &wT[n0 * KPAD_IN + kb + 16 + hi8]);
    acc0 = wmma_bf16f32(a, b0, acc0);
    v16bf b1 = load_frag(&wT[n1 * KPAD_IN + kb + hi8], &wT[n1 * KPAD_IN + kb + 16 + hi8]);
    acc1 = wmma_bf16f32(a, b1, acc1);
  }
  float bb0 = bias[n0], bb1 = bias[n1];
  int rofs = (lane >= 16) ? 8 : 0;
#pragma unroll
  for (int r = 0; r < 8; ++r) {
    int m = mbase + r + rofs;
    float v0 = acc0[r] + bb0; v0 = v0 > 0.f ? v0 : 0.f;
    float v1 = acc1[r] + bb1; v1 = v1 > 0.f ? v1 : 0.f;
    x0[m * HDIM + n0] = v0; x[m * HDIM + n0] = v0;
    x0[m * HDIM + n1] = v1; x[m * HDIM + n1] = v1;
  }
}

// -------- fused GCN2 layer: h=(1-a)agg+a*x0; y=(1-b)h+b*(h@W); x=relu(y+x) in place
__global__ void __launch_bounds__(128)
k_layer(const float* __restrict__ agg, const float* __restrict__ x0,
        float* __restrict__ x, const unsigned short* __restrict__ wTu, float beta) {
  const bf16* wT = (const bf16*)wTu;
  __shared__ __align__(16) bf16  sA[16 * HDIM];  // h in bf16 (WMMA A operand)
  __shared__ __align__(16) float sH[16 * HDIM];  // h in f32 (epilogue)
  int mbase = blockIdx.x * 16;
  int tid = threadIdx.x;
  for (int i = tid; i < 16 * HDIM; i += 128) {
    int g = (mbase + i / HDIM) * HDIM + (i % HDIM);
    float h = (1.0f - ALPHA_C) * agg[g] + ALPHA_C * x0[g];
    sH[i] = h;
    sA[i] = (bf16)h;
  }
  __syncthreads();
  int wave = tid >> 5, lane = tid & 31;
  int hi8  = (lane >= 16) ? 8  : 0;
  int hi16 = (lane >= 16) ? 16 : 0;
  int mrow = lane & 15;
  int n0 = (wave * 2 + 0) * 16 + mrow;
  int n1 = (wave * 2 + 1) * 16 + mrow;
  v8f acc0 = {}; v8f acc1 = {};
#pragma unroll
  for (int ks = 0; ks < HDIM / 32; ++ks) {
    int kb = ks * 32;
    v16bf a  = load_frag(&sA[mrow * HDIM + kb + hi8], &sA[mrow * HDIM + kb + 16 + hi8]);
    v16bf b0 = load_frag(&wT[n0 * HDIM + kb + hi8], &wT[n0 * HDIM + kb + 16 + hi8]);
    acc0 = wmma_bf16f32(a, b0, acc0);
    v16bf b1 = load_frag(&wT[n1 * HDIM + kb + hi8], &wT[n1 * HDIM + kb + 16 + hi8]);
    acc1 = wmma_bf16f32(a, b1, acc1);
  }
  int rofs = (lane >= 16) ? 8 : 0;
  float omb = 1.0f - beta;
#pragma unroll
  for (int r = 0; r < 8; ++r) {
    int mloc = r + rofs;
    int m = mbase + mloc;
    float y0 = omb * sH[mloc * HDIM + n0] + beta * acc0[r];
    float y1 = omb * sH[mloc * HDIM + n1] + beta * acc1[r];
    float v0 = y0 + x[m * HDIM + n0]; v0 = v0 > 0.f ? v0 : 0.f;
    float v1 = y1 + x[m * HDIM + n1]; v1 = v1 > 0.f ? v1 : 0.f;
    x[m * HDIM + n0] = v0;
    x[m * HDIM + n1] = v1;
  }
}

// -------- output projection: out = x @ w_out + b_out   [N,128]@[128,121]
__global__ void __launch_bounds__(128)
k_linear_out(const float* __restrict__ x, const unsigned short* __restrict__ wTu,
             const float* __restrict__ bias, float* __restrict__ out) {
  const bf16* wT = (const bf16*)wTu;
  __shared__ __align__(16) bf16 sA[16 * HDIM];
  int mbase = blockIdx.x * 16;
  int tid = threadIdx.x;
  for (int i = tid; i < 16 * HDIM; i += 128)
    sA[i] = (bf16)x[(mbase + i / HDIM) * HDIM + (i % HDIM)];
  __syncthreads();
  int wave = tid >> 5, lane = tid & 31;
  int hi8  = (lane >= 16) ? 8  : 0;
  int hi16 = (lane >= 16) ? 16 : 0;
  int mrow = lane & 15;
  int n0 = (wave * 2 + 0) * 16 + mrow;   // padded out-col
  int n1 = (wave * 2 + 1) * 16 + mrow;
  v8f acc0 = {}; v8f acc1 = {};
#pragma unroll
  for (int ks = 0; ks < HDIM / 32; ++ks) {
    int kb = ks * 32;
    v16bf a  = load_frag(&sA[mrow * HDIM + kb + hi8], &sA[mrow * HDIM + kb + 16 + hi8]);
    v16bf b0 = load_frag(&wT[n0 * HDIM + kb + hi8], &wT[n0 * HDIM + kb + 16 + hi8]);
    acc0 = wmma_bf16f32(a, b0, acc0);
    v16bf b1 = load_frag(&wT[n1 * HDIM + kb + hi8], &wT[n1 * HDIM + kb + 16 + hi8]);
    acc1 = wmma_bf16f32(a, b1, acc1);
  }
  float bb0 = (n0 < FOUT) ? bias[n0] : 0.0f;
  float bb1 = (n1 < FOUT) ? bias[n1] : 0.0f;
  int rofs = (lane >= 16) ? 8 : 0;
#pragma unroll
  for (int r = 0; r < 8; ++r) {
    int m = mbase + r + rofs;
    if (n0 < FOUT) out[m * FOUT + n0] = acc0[r] + bb0;
    if (n1 < FOUT) out[m * FOUT + n1] = acc1[r] + bb1;
  }
}

extern "C" void kernel_launch(void* const* d_in, const int* in_sizes, int n_in,
                              void* d_out, int out_size, void* d_ws, size_t ws_size,
                              hipStream_t stream) {
  const float*      x_in   = (const float*)d_in[0];      // [N, FIN]
  const long long*  eidx   = (const long long*)d_in[1];  // [2, E] int64
  const float*      ew     = (const float*)d_in[2];      // [E]
  const float*      w_in   = (const float*)d_in[3];      // [FIN, H]
  const float*      b_in   = (const float*)d_in[4];      // [H]
  const float*      conv_w = (const float*)d_in[5];      // [L, H, H]
  const float*      w_out  = (const float*)d_in[6];      // [H, FOUT]
  const float*      b_out  = (const float*)d_in[7];      // [FOUT]
  float*            out    = (float*)d_out;              // [N, FOUT]

  const size_t NH = (size_t)NNODES * HDIM;               // 6.4M floats
  char* ws = (char*)d_ws;
  float* x0  = (float*)(ws);
  float* xc  = (float*)(ws + NH * 4);
  float* agg = (float*)(ws + 2 * NH * 4);
  unsigned short* w_inT  = (unsigned short*)(ws + 3 * NH * 4);         // [128][64]  bf16
  unsigned short* convT  = w_inT  + 128 * KPAD_IN;                     // [6][128][128] bf16
  unsigned short* w_outT = convT  + NLAYERS * HDIM * HDIM;             // [128][128] bf16

  const long long* rowi = eidx;           // edge_index[0]
  const long long* coli = eidx + NEDGES;  // edge_index[1]

  // --- weights -> bf16, transposed (wT[n][k]) ---
  {
    int n = 128 * KPAD_IN;
    k_cvt_transpose<<<(n + 255) / 256, 256, 0, stream>>>(w_in, w_inT, FIN, HDIM, KPAD_IN, HDIM);
  }
  for (int l = 0; l < NLAYERS; ++l) {
    int n = HDIM * HDIM;
    k_cvt_transpose<<<(n + 255) / 256, 256, 0, stream>>>(
        conv_w + (size_t)l * HDIM * HDIM, convT + (size_t)l * HDIM * HDIM,
        HDIM, HDIM, HDIM, HDIM);
  }
  {
    int n = HDIM * HDIM;
    k_cvt_transpose<<<(n + 255) / 256, 256, 0, stream>>>(w_out, w_outT, HDIM, FOUT, HDIM, HDIM);
  }

  const int mblocks = NNODES / 16;  // 3125, exact

  // --- input projection ---
  k_linear_in<<<mblocks, 128, 0, stream>>>(x_in, w_inT, b_in, x0, xc);

  // --- 6 GCN2 layers ---
  const int n4 = (int)(NH / 4);
  const int spmm_threads = NEDGES * 32;
  for (int l = 0; l < NLAYERS; ++l) {
    float beta = logf(0.5f / (float)(l + 1) + 1.0f);
    k_zero4<<<(n4 + 255) / 256, 256, 0, stream>>>((float4*)agg, n4);
    k_spmm<<<(spmm_threads + 255) / 256, 256, 0, stream>>>(rowi, coli, ew, xc, agg);
    k_layer<<<mblocks, 128, 0, stream>>>(agg, x0, xc,
                                         convT + (size_t)l * HDIM * HDIM, beta);
  }

  // --- output projection ---
  k_linear_out<<<mblocks, 128, 0, stream>>>(xc, w_outT, b_out, out);
}
